// GNN_SchNet_3745211482667
// MI455X (gfx1250) — matrix operand order repaired and checked by reading, not compiled, not run on hardware
//
#include <hip/hip_runtime.h>

#define HD   128     // hidden / filter width
#define NI_C 6       // interactions
#define NG_C 50      // gaussians
#define NGP  64      // padded gaussians (WMMA K)

typedef __attribute__((ext_vector_type(16))) __bf16 v16bf;
typedef __attribute__((ext_vector_type(8)))  __bf16 v8bf;
typedef __attribute__((ext_vector_type(8)))  float  v8f;

union ABu { v16bf v; v8bf h[2]; };

__device__ __forceinline__ v8f wmma_bf16(v16bf a, v16bf b, v8f c) {
  return __builtin_amdgcn_wmma_f32_16x16x32_bf16(false, a, false, b, (short)0, c,
                                                 false, false);
}

// A fragment (16x32 bf16, row-major source, row = lr):
// g==0: halves[0..7]=K..K+7, halves[8..15]=K+16..K+23 ; g==1: +8 on both runs.
__device__ __forceinline__ v16bf load_a_bf16(const __bf16* base, int ldm, int row,
                                             int kb, int g) {
  ABu u;
  const __bf16* p = base + (size_t)row * ldm + kb + 8 * g;
  u.h[0] = *(const v8bf*)p;
  u.h[1] = *(const v8bf*)(p + 16);
  return u.v;
}

// B fragment (32x16 bf16) from transposed weights Bt[N][K]: col = lr,
// g==0: halves[0..7]=K..K+7, halves[8..15]=K+16..K+23 ; g==1: +8 on both runs.
__device__ __forceinline__ v16bf load_b_bf16(const __bf16* bt, int ldk, int col,
                                             int kb, int g) {
  ABu u;
  const __bf16* p = bt + (size_t)col * ldk + kb + 8 * g;
  u.h[0] = *(const v8bf*)p;
  u.h[1] = *(const v8bf*)(p + 16);
  return u.v;
}

// A fragment from fp32 source, converting to bf16 on the fly.
__device__ __forceinline__ v16bf load_a_f32(const float* base, int ldm, int row,
                                            int kb, int g) {
  const float* p = base + (size_t)row * ldm + kb + 8 * g;
  v16bf v;
#pragma unroll
  for (int t = 0; t < 8; ++t) v[t] = (__bf16)p[t];
#pragma unroll
  for (int t = 0; t < 8; ++t) v[8 + t] = (__bf16)p[16 + t];
  return v;
}

__device__ __forceinline__ float sspf(float x) {
  float sp = (x > 20.f) ? x : __logf(1.f + __expf(x));
  return sp - 0.69314718055994531f;
}

__device__ __forceinline__ void atomAddF(float* p, float v) {
  __hip_atomic_fetch_add(p, v, __ATOMIC_RELAXED, __HIP_MEMORY_SCOPE_AGENT);
}

// ---- CDNA5 async global->LDS staging (ASYNCcnt-tracked, no VGPR round-trip)
__device__ __forceinline__ unsigned lds_off_of(void* p) {
  return (unsigned)(uintptr_t)(__attribute__((address_space(3))) void*)p;
}
__device__ __forceinline__ void stage_async_b128(void* lds_dst,
                                                 const void* gsrc) {
  unsigned off = lds_off_of(lds_dst);
  asm volatile("global_load_async_to_lds_b128 %0, %1, off"
               :
               : "v"(off), "v"(gsrc)
               : "memory");
}
__device__ __forceinline__ void wait_async0() {
  asm volatile("s_wait_asynccnt 0" ::: "memory");
}
// stage 32 KB (128x128 bf16) weight matrix into LDS, 256 threads
__device__ __forceinline__ void stage_weights(v8bf* dst, const v8bf* src) {
#pragma unroll
  for (int it = 0; it < (HD * HD / 8) / 256; ++it) {
    int i = it * 256 + threadIdx.x;
    stage_async_b128(dst + i, src + i);
  }
  wait_async0();
  __syncthreads();
}

// ---------------------------------------------------------------- utilities
__global__ void fill_zero(float* p, long n) {
  long i = (long)blockIdx.x * blockDim.x + threadIdx.x;
  if (i < n) p[i] = 0.f;
}

// src: [count][K][Nout] fp32  ->  dst: [count][Nout][Kpad] bf16 (zero pad K)
__global__ void transpose_w(const float* src, __bf16* dst, int K, int Kpad,
                            int Nout, int count) {
  long idx = (long)blockIdx.x * blockDim.x + threadIdx.x;
  long total = (long)count * Nout * Kpad;
  if (idx >= total) return;
  int k = idx % Kpad;
  long t = idx / Kpad;
  int n = t % Nout;
  int i = t / Nout;
  float v = (k < K) ? src[((long)i * K + k) * Nout + n] : 0.f;
  dst[idx] = (__bf16)v;
}

__global__ void embed_kernel(const float* emb, const int* z, float* h,
                             __bf16* hb, long total) {
  long idx = (long)blockIdx.x * blockDim.x + threadIdx.x;
  if (idx >= total) return;
  long n = idx >> 7;
  int k = idx & (HD - 1);
  float v = emb[(long)z[n] * HD + k];
  h[idx] = v;
  hb[idx] = (__bf16)v;
}

// ------------------------------------------------------------- edge geometry
__global__ void edge_geom(const float* pos, const int* row, const int* col,
                          __bf16* rbfb, float* Cc, int E) {
  long e = (long)blockIdx.x * blockDim.x + threadIdx.x;
  if (e >= E) return;
  int i = row[e], j = col[e];
  float dx = pos[3 * (long)i + 0] - pos[3 * (long)j + 0];
  float dy = pos[3 * (long)i + 1] - pos[3 * (long)j + 1];
  float dz = pos[3 * (long)i + 2] - pos[3 * (long)j + 2];
  float d = __builtin_sqrtf(dx * dx + dy * dy + dz * dz + 1e-12f);
  Cc[e] = 0.5f * (__cosf(d * (3.14159265358979f / 10.f)) + 1.f);
  const float delta = 10.f / 49.f;
  const float coeff = -0.5f / (delta * delta);
#pragma unroll
  for (int gq = 0; gq < NGP; ++gq) {
    float v = 0.f;
    if (gq < NG_C) {
      float t = d - (float)gq * delta;
      v = __expf(coeff * t * t);
    }
    rbfb[e * (long)NGP + gq] = (__bf16)v;
  }
}

// ------------------------------------------------------- x = h @ lin1 (WMMA)
// Dynamic LDS: staged Bt (128x128 bf16 = 32 KB), filled by async engine.
__global__ void __launch_bounds__(256) node_gemm(const __bf16* Abf,
                                                 const __bf16* Bt, __bf16* outb,
                                                 int N) {
  extern __shared__ __bf16 dsm[];
  stage_weights((v8bf*)dsm, (const v8bf*)Bt);

  int w = threadIdx.x >> 5, lane = threadIdx.x & 31;
  int g = lane >> 4, lr = lane & 15;
  long tile = (long)blockIdx.x * 8 + w;
  long base = tile * 16;
  if (base >= N) return;
  bool full = (base + 16 <= (long)N);

  int arow = (int)((base + lr < N) ? base + lr : N - 1);
  v16bf a[4];
#pragma unroll
  for (int ks = 0; ks < 4; ++ks) a[ks] = load_a_bf16(Abf, HD, arow, ks * 32, g);
#pragma unroll
  for (int nt = 0; nt < 8; ++nt) {
    v8f acc = {};
#pragma unroll
    for (int ks = 0; ks < 4; ++ks)
      acc = wmma_bf16(a[ks], load_b_bf16(dsm, HD, nt * 16 + lr, ks * 32, g), acc);
    int feat = nt * 16 + lr;
    if (full) {
#pragma unroll
      for (int r = 0; r < 8; ++r)
        outb[(base + 8 * g + r) * HD + feat] = (__bf16)acc[r];
    } else {
#pragma unroll
      for (int r = 0; r < 8; ++r) {
        long node = base + 8 * g + r;
        if (node < N) outb[node * HD + feat] = (__bf16)acc[r];
      }
    }
  }
}

// --------------- fused: W = ssp(rbf@w1+b1)@w2+b2, *C, gather x[col], scatter
// Dynamic LDS: w2s (32 KB, async-staged) + 8 per-wave tiles (32 KB) = 64 KB.
__global__ void __launch_bounds__(256) edge_filter_scatter(
    const __bf16* rbfb, const float* Cc, const __bf16* w1t, const float* b1,
    const __bf16* w2t, const float* b2, const __bf16* xb, const int* rowI,
    const int* colI, float* agg, int E) {
  extern __shared__ __bf16 dsm[];
  __bf16* w2s = dsm;                 // [128][128]
  __bf16* tiles = dsm + HD * HD;     // 8 x [16][128]
  stage_weights((v8bf*)w2s, (const v8bf*)w2t);

  int w = threadIdx.x >> 5, lane = threadIdx.x & 31;
  int g = lane >> 4, lr = lane & 15;
  __bf16* tl = tiles + (size_t)w * 16 * HD;
  long tile = (long)blockIdx.x * 8 + w;
  long base = tile * 16;
  if (base >= E) return;
  bool full = (base + 16 <= (long)E);

  // per-output-row edge metadata (M = 8g + r), prefetch gather rows early
  int cn[8], rn[8];
  float cv[8];
#pragma unroll
  for (int r = 0; r < 8; ++r) {
    long e = base + 8 * g + r;
    long ec = (e < E) ? e : 0;
    cn[r] = colI[ec];
    rn[r] = rowI[ec];
    cv[r] = Cc[ec];
  }
#pragma unroll
  for (int r = 0; r < 8; ++r)
    __builtin_prefetch(xb + (long)cn[r] * HD, 0, 3);  // global_prefetch_b8

  // GEMM1: t[16,128] = ssp(rbf[16,64] @ w1 + b1) -> LDS (row-major bf16)
  int arow = (int)((base + lr < E) ? base + lr : E - 1);
  v16bf a0 = load_a_bf16(rbfb, NGP, arow, 0, g);
  v16bf a1 = load_a_bf16(rbfb, NGP, arow, 32, g);
#pragma unroll
  for (int nt = 0; nt < 8; ++nt) {
    float bias = b1[nt * 16 + lr];
    v8f acc = {};
    acc = wmma_bf16(a0, load_b_bf16(w1t, NGP, nt * 16 + lr, 0, g), acc);
    acc = wmma_bf16(a1, load_b_bf16(w1t, NGP, nt * 16 + lr, 32, g), acc);
#pragma unroll
    for (int r = 0; r < 8; ++r)
      tl[(8 * g + r) * HD + nt * 16 + lr] = (__bf16)sspf(acc[r] + bias);
  }
  asm volatile("" ::: "memory");  // wave-private LDS tile; LDS is in-order

  // GEMM2 A fragments from LDS
  v16bf ta[4];
#pragma unroll
  for (int ks = 0; ks < 4; ++ks) ta[ks] = load_a_bf16(tl, HD, lr, ks * 32, g);

#pragma unroll
  for (int nt = 0; nt < 8; ++nt) {
    float bias2 = b2[nt * 16 + lr];
    v8f acc = {};
#pragma unroll
    for (int ks = 0; ks < 4; ++ks)
      acc = wmma_bf16(ta[ks], load_b_bf16(w2s, HD, nt * 16 + lr, ks * 32, g), acc);
    int feat = nt * 16 + lr;
    if (full) {
#pragma unroll
      for (int r = 0; r < 8; ++r) {
        float Wv = (acc[r] + bias2) * cv[r];
        float xv = (float)xb[(long)cn[r] * HD + feat];
        atomAddF(&agg[(long)rn[r] * HD + feat], Wv * xv);
      }
    } else {
#pragma unroll
      for (int r = 0; r < 8; ++r) {
        if (base + 8 * g + r >= E) continue;
        float Wv = (acc[r] + bias2) * cv[r];
        float xv = (float)xb[(long)cn[r] * HD + feat];
        atomAddF(&agg[(long)rn[r] * HD + feat], Wv * xv);
      }
    }
  }
}

// ----------- fused: h += (ssp(agg@lin2+b) @ lin_w + b); emit h fp32 + bf16
// Dynamic LDS: lin2s (32 KB, async-staged) + 8 per-wave tiles (32 KB) = 64 KB.
__global__ void __launch_bounds__(256) node_update(
    const float* agg, const __bf16* lin2t, const float* lin2_b,
    const __bf16* lint, const float* lin_b, float* h, __bf16* hb, int N) {
  extern __shared__ __bf16 dsm[];
  __bf16* l2s = dsm;                 // [128][128]
  __bf16* tiles = dsm + HD * HD;     // 8 x [16][128]
  stage_weights((v8bf*)l2s, (const v8bf*)lin2t);

  int w = threadIdx.x >> 5, lane = threadIdx.x & 31;
  int g = lane >> 4, lr = lane & 15;
  __bf16* tl = tiles + (size_t)w * 16 * HD;
  long tile = (long)blockIdx.x * 8 + w;
  long base = tile * 16;
  if (base >= N) return;
  bool full = (base + 16 <= (long)N);

  int arow = (int)((base + lr < N) ? base + lr : N - 1);
  v16bf a[4];
#pragma unroll
  for (int ks = 0; ks < 4; ++ks) a[ks] = load_a_f32(agg, HD, arow, ks * 32, g);
#pragma unroll
  for (int nt = 0; nt < 8; ++nt) {
    float bias = lin2_b[nt * 16 + lr];
    v8f acc = {};
#pragma unroll
    for (int ks = 0; ks < 4; ++ks)
      acc = wmma_bf16(a[ks], load_b_bf16(l2s, HD, nt * 16 + lr, ks * 32, g), acc);
#pragma unroll
    for (int r = 0; r < 8; ++r)
      tl[(8 * g + r) * HD + nt * 16 + lr] = (__bf16)sspf(acc[r] + bias);
  }
  asm volatile("" ::: "memory");

  v16bf ta[4];
#pragma unroll
  for (int ks = 0; ks < 4; ++ks) ta[ks] = load_a_bf16(tl, HD, lr, ks * 32, g);
#pragma unroll
  for (int nt = 0; nt < 8; ++nt) {
    int feat = nt * 16 + lr;
    float bias = lin_b[feat];
    v8f acc = {};
#pragma unroll
    for (int ks = 0; ks < 4; ++ks)
      acc = wmma_bf16(ta[ks], load_b_bf16(lint, HD, nt * 16 + lr, ks * 32, g), acc);
    if (full) {
#pragma unroll
      for (int r = 0; r < 8; ++r) {
        long o = (base + 8 * g + r) * HD + feat;
        float hn = h[o] + acc[r] + bias;
        h[o] = hn;
        hb[o] = (__bf16)hn;
      }
    } else {
#pragma unroll
      for (int r = 0; r < 8; ++r) {
        long node = base + 8 * g + r;
        if (node < N) {
          long o = node * HD + feat;
          float hn = h[o] + acc[r] + bias;
          h[o] = hn;
          hb[o] = (__bf16)hn;
        }
      }
    }
  }
}

// ----------------------------------------------------------------- readout
__global__ void readout(const float* h, const float* ow1, const float* ob1,
                        const float* ow2, const float* ob2, const int* batch,
                        float* energy, int N) {
  long n = (long)blockIdx.x * blockDim.x + threadIdx.x;
  if (n >= N) return;
  const float* hr = h + n * HD;
  float e = ob2[0];
#pragma unroll 1
  for (int j = 0; j < HD / 2; ++j) {
    float s = ob1[j];
#pragma unroll 4
    for (int k = 0; k < HD; ++k) s += hr[k] * ow1[k * (HD / 2) + j];
    e += sspf(s) * ow2[j];
  }
  atomAddF(&energy[batch[n]], e);
}

// ------------------------------------------------------------------ launcher
extern "C" void kernel_launch(void* const* d_in, const int* in_sizes, int n_in,
                              void* d_out, int out_size, void* d_ws,
                              size_t ws_size, hipStream_t stream) {
  const float* pos    = (const float*)d_in[0];
  const float* emb    = (const float*)d_in[1];
  const float* w1     = (const float*)d_in[2];
  const float* b1     = (const float*)d_in[3];
  const float* w2     = (const float*)d_in[4];
  const float* b2     = (const float*)d_in[5];
  const float* lin1w  = (const float*)d_in[6];
  const float* lin2w  = (const float*)d_in[7];
  const float* lin2b  = (const float*)d_in[8];
  const float* linw   = (const float*)d_in[9];
  const float* linb   = (const float*)d_in[10];
  const float* ow1    = (const float*)d_in[11];
  const float* ob1    = (const float*)d_in[12];
  const float* ow2    = (const float*)d_in[13];
  const float* ob2    = (const float*)d_in[14];
  const int*   z      = (const int*)d_in[15];
  const int*   batch  = (const int*)d_in[16];
  const int*   eidx   = (const int*)d_in[17];

  const int N = in_sizes[15];
  const int E = in_sizes[17] / 2;
  const int G = out_size;
  const int* rowI = eidx;
  const int* colI = eidx + E;

  // workspace carving (256B aligned)
  char* ws = (char*)d_ws;
  size_t off = 0;
  auto carve = [&](size_t bytes) {
    char* p = ws + off;
    off += (bytes + 255) & ~(size_t)255;
    return p;
  };
  __bf16* rbfb = (__bf16*)carve((size_t)E * NGP * 2);
  float*  Cc   = (float*)carve((size_t)E * 4);
  float*  h    = (float*)carve((size_t)N * HD * 4);
  __bf16* hb   = (__bf16*)carve((size_t)N * HD * 2);
  __bf16* xb   = (__bf16*)carve((size_t)N * HD * 2);
  float*  agg  = (float*)carve((size_t)N * HD * 4);
  __bf16* w1t  = (__bf16*)carve((size_t)NI_C * HD * NGP * 2);
  __bf16* w2t  = (__bf16*)carve((size_t)NI_C * HD * HD * 2);
  __bf16* l1t  = (__bf16*)carve((size_t)NI_C * HD * HD * 2);
  __bf16* l2t  = (__bf16*)carve((size_t)NI_C * HD * HD * 2);
  __bf16* lwt  = (__bf16*)carve((size_t)NI_C * HD * HD * 2);
  (void)ws_size;

  const int T = 256;
  auto blk = [](long n, int t) { return (unsigned)((n + t - 1) / t); };

  // weight prep (bf16, transposed to [out][K])
  transpose_w<<<blk((long)NI_C * HD * NGP, T), T, 0, stream>>>(w1, w1t, NG_C, NGP, HD, NI_C);
  transpose_w<<<blk((long)NI_C * HD * HD, T), T, 0, stream>>>(w2, w2t, HD, HD, HD, NI_C);
  transpose_w<<<blk((long)NI_C * HD * HD, T), T, 0, stream>>>(lin1w, l1t, HD, HD, HD, NI_C);
  transpose_w<<<blk((long)NI_C * HD * HD, T), T, 0, stream>>>(lin2w, l2t, HD, HD, HD, NI_C);
  transpose_w<<<blk((long)NI_C * HD * HD, T), T, 0, stream>>>(linw, lwt, HD, HD, HD, NI_C);

  edge_geom<<<blk(E, T), T, 0, stream>>>(pos, rowI, colI, rbfb, Cc, E);
  embed_kernel<<<blk((long)N * HD, T), T, 0, stream>>>(emb, z, h, hb, (long)N * HD);

  const unsigned nodeBlocks = blk(((long)N + 15) / 16, 8);
  const unsigned edgeBlocks = blk(((long)E + 15) / 16, 8);
  const size_t shW  = (size_t)HD * HD * 2;             // 32 KB staged weights
  const size_t shWT = shW + (size_t)8 * 16 * HD * 2;   // + 32 KB wave tiles

  for (int i = 0; i < NI_C; ++i) {
    fill_zero<<<blk((long)N * HD, T), T, 0, stream>>>(agg, (long)N * HD);
    node_gemm<<<nodeBlocks, T, shW, stream>>>(hb, l1t + (size_t)i * HD * HD, xb, N);
    edge_filter_scatter<<<edgeBlocks, T, shWT, stream>>>(
        rbfb, Cc, w1t + (size_t)i * HD * NGP, b1 + (size_t)i * HD,
        w2t + (size_t)i * HD * HD, b2 + (size_t)i * HD, xb, rowI, colI, agg, E);
    node_update<<<nodeBlocks, T, shWT, stream>>>(
        agg, l2t + (size_t)i * HD * HD, lin2b + (size_t)i * HD,
        lwt + (size_t)i * HD * HD, linb + (size_t)i * HD, h, hb, N);
  }

  fill_zero<<<blk(G, T), T, 0, stream>>>((float*)d_out, G);
  readout<<<blk(N, T), T, 0, stream>>>(h, ow1, ob1, ow2, ob2, batch,
                                       (float*)d_out, N);
}
